// LinearAttention_3006477107401
// MI455X (gfx1250) — matrix operand order repaired and checked by reading, not compiled, not run on hardware
//
#include <hip/hip_runtime.h>
#include <hip/hip_bf16.h>

// ---------------------------------------------------------------------------
// CDNA5 / gfx1250 WMMA types + helpers (wave32).
// All matrix operands are kept in "fragment layout" in memory so that a
// fragment load is one contiguous 32-byte read (2x b128) per lane.
//
// A-layout (16x32, 16-bit):  lane = m | ((k&8)<<1) ; slot = ((k&16)>>1)|(k&7)
// B-layout (32x16, 16-bit):  lane = n | ((k&8)<<1) ; slot = ((k&16)>>1)|(k&7)
// C/D frag: VGPR g, lane ->  row M = g + (lane>=16 ? 8:0), col N = lane&15
// ---------------------------------------------------------------------------
typedef __attribute__((ext_vector_type(16))) __bf16 bf16x16;
typedef __attribute__((ext_vector_type(8)))  float  f32x8;

__device__ __forceinline__ f32x8 wmma_bf(bf16x16 a, bf16x16 b, f32x8 c) {
  return __builtin_amdgcn_wmma_f32_16x16x32_bf16(
      /*neg_a=*/false, a, /*neg_b=*/false, b,
      /*c_mod=*/(short)0, c, /*reuse_a=*/false, /*reuse_b=*/false);
}

__device__ __forceinline__ bf16x16 frag32(const __bf16* p) {
  return *(const bf16x16*)p;  // 32B contiguous -> 2x load_b128
}

// ---------------------------------------------------------------------------
// Problem constants
// ---------------------------------------------------------------------------
#define BB     2
#define SS     4096
#define EE     512
#define HH     16
#define DD     32
#define CCHUNK 128
#define NCHUNK 32
#define TILE_E 512          // elements per 16x32 / 32x16 fragment tile

// ---------------------------------------------------------------------------
// Prep: x (f32 row-major [8192][512]) -> bf16 A-layout tiles
//       x_sw[(rowtile*16 + ktile)*512 + lane*16 + slot]
// ---------------------------------------------------------------------------
__global__ __launch_bounds__(256) void prep_x(const float* __restrict__ x,
                                              __bf16* __restrict__ xsw) {
  const int idx = (blockIdx.x * 256 + threadIdx.x) * 4;  // 4M elems, grid 4096
  const float4 v = *(const float4*)(x + idx);
  const int m = idx >> 9;
  const int k = idx & 511;
  const float f[4] = {v.x, v.y, v.z, v.w};
#pragma unroll
  for (int j = 0; j < 4; ++j) {
    const int kk = k + j, k2 = kk & 31;
    const int lane2 = (m & 15) | ((k2 & 8) << 1);
    const int slot2 = ((k2 & 16) >> 1) | (k2 & 7);
    xsw[(size_t)((m >> 4) * 16 + (kk >> 5)) * TILE_E + lane2 * 16 + slot2] = (__bf16)f[j];
  }
}

// ---------------------------------------------------------------------------
// Prep: Wq/Wk/Wv/Wo (f32 [512][512]) -> bf16 B-layout tiles
//       w_sw[mat*256K + (ktile*32 + ntile)*512 + lane*16 + slot]
// ---------------------------------------------------------------------------
__global__ __launch_bounds__(256) void prep_w(const float* __restrict__ Wq,
                                              const float* __restrict__ Wk,
                                              const float* __restrict__ Wv,
                                              const float* __restrict__ Wo,
                                              __bf16* __restrict__ wsw) {
  const int mat = blockIdx.y;
  const float* W = (mat == 0) ? Wq : (mat == 1) ? Wk : (mat == 2) ? Wv : Wo;
  const int idx = (blockIdx.x * 256 + threadIdx.x) * 4;  // 256K per mat, grid.x 256
  const float4 v = *(const float4*)(W + idx);
  const int k = idx >> 9, n = idx & 511, k2 = k & 31;
  const float f[4] = {v.x, v.y, v.z, v.w};
  __bf16* dst = wsw + (size_t)mat * (512 * 512);
#pragma unroll
  for (int j = 0; j < 4; ++j) {
    const int nj = n + j;
    const int lane2 = (nj & 15) | ((k2 & 8) << 1);
    dst[(size_t)((k >> 5) * 32 + (nj >> 4)) * TILE_E + lane2 * 16 + (((k2 & 16) >> 1) | (k2 & 7))] = (__bf16)f[j];
  }
}

// ---------------------------------------------------------------------------
// Kernel 1: QKV projection (LDS-free; frags straight from L2).
// Writes Q/K/V pre-swizzled for the attention kernel:
//   qg : per (bh,chunk) 8 strips  - A-layout of Q (k = d)
//   kbg: per (bh,chunk) 8 c-tiles - B-layout of K^T (B[d][c])
//   kag: per (bh,chunk) ti*4+kt   - A-layout of K^T (A[d][c])
//   vg : per (bh,chunk) kt*2+tj   - B-layout of V  (B[c][d])
// ---------------------------------------------------------------------------
__global__ __launch_bounds__(256) void qkv_kernel(
    const __bf16* __restrict__ xsw, const __bf16* __restrict__ wsw,
    const float* __restrict__ bq, const float* __restrict__ bk,
    const float* __restrict__ bv,
    __bf16* __restrict__ qg, __bf16* __restrict__ kbg,
    __bf16* __restrict__ kag, __bf16* __restrict__ vg) {
  const int mt = blockIdx.x, ntb = blockIdx.y, mat = blockIdx.z;
  const __bf16* W = wsw + (size_t)mat * (512 * 512);
  const float* bias = (mat == 0) ? bq : (mat == 1) ? bk : bv;
  const int tid = threadIdx.x, w = tid >> 5, lane = tid & 31;
  const int hi = lane >> 4, nn = lane & 15;
  const int rt = mt * 8 + w;

  f32x8 acc[8] = {};
  for (int kt = 0; kt < 16; ++kt) {
    const bf16x16 a = frag32(xsw + (size_t)(rt * 16 + kt) * TILE_E + lane * 16);
#pragma unroll
    for (int t = 0; t < 8; ++t) {
      const bf16x16 b = frag32(W + (size_t)(kt * 32 + ntb * 8 + t) * TILE_E + lane * 16);
      acc[t] = wmma_bf(a, b, acc[t]);
    }
  }

#pragma unroll
  for (int t = 0; t < 8; ++t) {
#pragma unroll
    for (int g = 0; g < 8; ++g) {
      const int m = mt * 128 + w * 16 + g + (hi ? 8 : 0);
      const int n = ntb * 128 + t * 16 + nn;
      float v = acc[t][g] + bias[n];
      if (mat < 2) v = (v > 0.f) ? (v + 1.f) : __expf(v);  // phi = elu+1
      const int bidx = m >> 12, s = m & 4095, c = s >> 7, r = s & 127;
      const int h = n >> 5, d = n & 31;
      const size_t cb = (size_t)((bidx * HH + h) * NCHUNK + c) * 4096;
      const __bf16 bv16 = (__bf16)v;
      if (mat == 0) {        // Q: A-layout, strip = r>>4, k = d
        const int l2 = (r & 15) | ((d & 8) << 1);
        const int s2 = ((d & 16) >> 1) | (d & 7);
        qg[cb + (size_t)(r >> 4) * TILE_E + l2 * 16 + s2] = bv16;
      } else if (mat == 1) { // K: both B-layout (Q.K^T) and A-layout (K^T.V)
        kbg[cb + (size_t)(r >> 4) * TILE_E + ((r & 15) | ((d & 8) << 1)) * 16 + (((d & 16) >> 1) | (d & 7))] = bv16;
        const int l2 = (d & 15) | ((r & 8) << 1);
        const int s2 = ((r & 16) >> 1) | (r & 7);
        kag[cb + (size_t)((d >> 4) * 4 + (r >> 5)) * TILE_E + l2 * 16 + s2] = bv16;
      } else {               // V: B-layout, tile = kt*2+tj
        vg[cb + (size_t)((r >> 5) * 2 + (d >> 4)) * TILE_E +
           ((d & 15) | ((r & 8) << 1)) * 16 + (((r & 16) >> 1) | (r & 7))] = bv16;
      }
    }
  }
}

// ---------------------------------------------------------------------------
// Kernel 2: chunked causal linear-attention scan, one block per (b,h).
// V' = [V | 1] and S' = [S | z] so rowsum(A)+Q.z (the denominator) is just a
// third WMMA N-tile; no LDS atomics / scalar reductions needed.
// Result written directly in oproj's A-fragment layout (rsw).
// ---------------------------------------------------------------------------
__global__ __launch_bounds__(256) void attn_kernel(
    const __bf16* __restrict__ qg, const __bf16* __restrict__ kbg,
    const __bf16* __restrict__ kag, const __bf16* __restrict__ vg,
    __bf16* __restrict__ rsw) {
  __shared__ __bf16 Asw[8 * 4 * TILE_E];  // masked A, A-layout, per-strip  (32KB)
  __shared__ float  Stf[32 * 48];         // f32 master of S' = [S | z]     (6KB)
  __shared__ __bf16 Ssw[3 * TILE_E];      // bf16 B-layout copy of S'       (3KB)

  const int tid = threadIdx.x, w = tid >> 5, lane = tid & 31;
  const int hi = lane >> 4, nn = lane & 15;
  const int bh = blockIdx.x, b = bh >> 4, h = bh & 15;

  for (int i = tid; i < 32 * 48; i += 256) Stf[i] = 0.f;
  for (int i = tid; i < 3 * TILE_E; i += 256) Ssw[i] = (__bf16)0.f;
  __syncthreads();

  // Constant B-fragment of the all-[col0 = 1] matrix (ones column of V').
  bf16x16 ones;
#pragma unroll
  for (int i = 0; i < 16; ++i) ones[i] = (nn == 0) ? (__bf16)1.f : (__bf16)0.f;

  for (int c = 0; c < NCHUNK; ++c) {
    const size_t cb = (size_t)(bh * NCHUNK + c) * 4096;

    // ---- phase 1: A = tril(Qc Kc^T) for own 16-row strip -> Asw ----------
    const bf16x16 qa = frag32(qg + cb + (size_t)w * TILE_E + lane * 16);
#pragma unroll
    for (int t = 0; t < 8; ++t) {
      f32x8 az = {};
      if (t <= w) {  // wave-uniform: tiles above the diagonal are all-zero
        const bf16x16 kf = frag32(kbg + cb + (size_t)t * TILE_E + lane * 16);
        az = wmma_bf(qa, kf, az);
      }
#pragma unroll
      for (int g = 0; g < 8; ++g) {
        const int m2 = g + (hi ? 8 : 0);
        const int row = w * 16 + m2, col = t * 16 + nn, k2 = col & 31;
        const float v = (col <= row) ? az[g] : 0.f;  // causal mask
        Asw[(w * 4 + (col >> 5)) * TILE_E + ((m2 | ((k2 & 8) << 1)) << 4) +
            (((k2 & 16) >> 1) | (k2 & 7))] = (__bf16)v;
      }
    }

    // ---- phase 2: [num|den] = A V' + Qc S'_old ; out = num/(den+eps) -----
    // (same-wave DS write->read on Asw: in-order per ISA, no barrier needed)
    f32x8 o0 = {}, o1 = {}, o2 = {};
#pragma unroll
    for (int kt = 0; kt < 4; ++kt) {
      const bf16x16 af = frag32(Asw + (w * 4 + kt) * TILE_E + lane * 16);
      const bf16x16 v0 = frag32(vg + cb + (size_t)(kt * 2 + 0) * TILE_E + lane * 16);
      const bf16x16 v1 = frag32(vg + cb + (size_t)(kt * 2 + 1) * TILE_E + lane * 16);
      o0 = wmma_bf(af, v0, o0);
      o1 = wmma_bf(af, v1, o1);
      o2 = wmma_bf(af, ones, o2);
    }
    {
      const bf16x16 s0 = frag32(Ssw + 0 * TILE_E + lane * 16);
      const bf16x16 s1 = frag32(Ssw + 1 * TILE_E + lane * 16);
      const bf16x16 s2 = frag32(Ssw + 2 * TILE_E + lane * 16);
      o0 = wmma_bf(qa, s0, o0);
      o1 = wmma_bf(qa, s1, o1);
      o2 = wmma_bf(qa, s2, o2);
    }
#pragma unroll
    for (int g = 0; g < 8; ++g) {
      const int m2 = g + (hi ? 8 : 0);
      // den lives in column n'=0 of o2 (lanes 0 / 16); broadcast to the half
      const float dd = __shfl(o2[g], lane & 16, 32) + 1e-6f;
      const float rd = 1.f / dd;
      // store in oproj A-layout: rowtile = b*256 + c*8 + w, ktile = h
      const int l2 = m2 | ((nn & 8) << 1);
      const size_t base2 =
          (size_t)(((b * 256 + c * 8 + w) * 16 + h)) * TILE_E + l2 * 16;
      rsw[base2 + (nn & 7)]     = (__bf16)(o0[g] * rd);  // cols h*32 + nn
      rsw[base2 + 8 + (nn & 7)] = (__bf16)(o1[g] * rd);  // cols h*32+16+nn
    }
    __syncthreads();  // all S' readers done before update

    // ---- phase 3: S' += Kc^T V'  (6 tiles on waves 0..5) ------------------
    if (w < 6) {  // wave-uniform branch: EXEC all-1s inside
      const int ti = w / 3, tj = w % 3;
      f32x8 sacc = {};
#pragma unroll
      for (int kt = 0; kt < 4; ++kt) {
        const bf16x16 ka = frag32(kag + cb + (size_t)(ti * 4 + kt) * TILE_E + lane * 16);
        bf16x16 bb;
        if (tj < 2) bb = frag32(vg + cb + (size_t)(kt * 2 + tj) * TILE_E + lane * 16);
        else        bb = ones;  // ones column -> z update
        sacc = wmma_bf(ka, bb, sacc);
      }
#pragma unroll
      for (int g = 0; g < 8; ++g) {
        const int d = ti * 16 + g + (hi ? 8 : 0);
        const int ci = d * 48 + tj * 16 + nn;
        const float ns = Stf[ci] + sacc[g];
        Stf[ci] = ns;  // f32 master carries precision across chunks
        Ssw[tj * TILE_E + ((nn | ((d & 8) << 1)) << 4) + (((d & 16) >> 1) | (d & 7))] = (__bf16)ns;
      }
    }
    __syncthreads();
  }
}

// ---------------------------------------------------------------------------
// Kernel 3: output projection out = res @ Wo + bo (LDS-free, f32 out).
// ---------------------------------------------------------------------------
__global__ __launch_bounds__(256) void oproj_kernel(
    const __bf16* __restrict__ rsw, const __bf16* __restrict__ wo_sw,
    const float* __restrict__ bo, float* __restrict__ out) {
  const int mt = blockIdx.x, ntb = blockIdx.y;
  const int tid = threadIdx.x, w = tid >> 5, lane = tid & 31;
  const int hi = lane >> 4, nn = lane & 15;
  const int rt = mt * 8 + w;

  f32x8 acc[8] = {};
  for (int kt = 0; kt < 16; ++kt) {
    const bf16x16 a = frag32(rsw + (size_t)(rt * 16 + kt) * TILE_E + lane * 16);
#pragma unroll
    for (int t = 0; t < 8; ++t) {
      const bf16x16 b = frag32(wo_sw + (size_t)(kt * 32 + ntb * 8 + t) * TILE_E + lane * 16);
      acc[t] = wmma_bf(a, b, acc[t]);
    }
  }
#pragma unroll
  for (int t = 0; t < 8; ++t) {
#pragma unroll
    for (int g = 0; g < 8; ++g) {
      const int m = mt * 128 + w * 16 + g + (hi ? 8 : 0);
      const int n = ntb * 128 + t * 16 + nn;
      out[(size_t)m * EE + n] = acc[t][g] + bo[n];
    }
  }
}

// ---------------------------------------------------------------------------
// Launch
// ---------------------------------------------------------------------------
extern "C" void kernel_launch(void* const* d_in, const int* in_sizes, int n_in,
                              void* d_out, int out_size, void* d_ws, size_t ws_size,
                              hipStream_t stream) {
  (void)in_sizes; (void)n_in; (void)out_size; (void)ws_size;
  const float* x  = (const float*)d_in[0];
  const float* Wq = (const float*)d_in[1];
  const float* bq = (const float*)d_in[2];
  const float* Wk = (const float*)d_in[3];
  const float* bk = (const float*)d_in[4];
  const float* Wv = (const float*)d_in[5];
  const float* bv = (const float*)d_in[6];
  const float* Wo = (const float*)d_in[7];
  const float* bo = (const float*)d_in[8];
  float* out = (float*)d_out;

  const size_t QN = (size_t)BB * HH * SS * DD;  // 4,194,304 elems per tensor
  __bf16* xsw = (__bf16*)d_ws;                  // also reused as rsw (res)
  __bf16* wsw = xsw + QN;                       // 4 * 512*512
  __bf16* qg  = wsw + (size_t)4 * 512 * 512;
  __bf16* kbg = qg + QN;
  __bf16* kag = kbg + QN;
  __bf16* vgp = kag + QN;
  __bf16* rsw = xsw;  // x is dead after qkv_kernel; alias for res

  prep_x<<<dim3(4096), 256, 0, stream>>>(x, xsw);
  prep_w<<<dim3(256, 4), 256, 0, stream>>>(Wq, Wk, Wv, Wo, wsw);
  qkv_kernel<<<dim3(64, 4, 3), 256, 0, stream>>>(xsw, wsw, bq, bk, bv,
                                                 qg, kbg, kag, vgp);
  attn_kernel<<<dim3(BB * HH), 256, 0, stream>>>(qg, kbg, kag, vgp, rsw);
  oproj_kernel<<<dim3(64, 4), 256, 0, stream>>>(rsw, wsw + (size_t)3 * 512 * 512,
                                                bo, out);
}
